// MultiHeadAttentionWithRelativePosition_5875515261489
// MI455X (gfx1250) — matrix operand order repaired and checked by reading, not compiled, not run on hardware
//
#include <hip/hip_runtime.h>
#include <hip/hip_bf16.h>
#include <math.h>

// ---------------- problem constants ----------------
#define D_MODEL   512
#define NUM_HEADS 8
#define HEAD_DIM  64
#define MAX_DIST  60
#define BATCH     8
#define SEQ       2048
#define NROW      (BATCH * SEQ)          // 16384 rows of x

typedef __attribute__((ext_vector_type(16))) __bf16          v16bf;
typedef __attribute__((ext_vector_type(16))) unsigned short  us16;
typedef __attribute__((ext_vector_type(8)))  float           v8f;

#define WMMA_BF16(a, b, c) \
  __builtin_amdgcn_wmma_f32_16x16x32_bf16(false, (a), false, (b), (short)0, (c), false, false)

// f32 -> bf16 round-to-nearest-even
__device__ __forceinline__ unsigned short f2bf(float f) {
  unsigned int u = __float_as_uint(f);
  unsigned int r = (u + 0x7FFFu + ((u >> 16) & 1u)) >> 16;
  return (unsigned short)r;
}

// ---- CDNA5 async global->LDS copy (16B per lane), tracked by ASYNCcnt ----
__device__ __forceinline__ void async_g2l_b128(void* lds_dst, const void* gsrc) {
  unsigned int lds_addr = (unsigned int)(size_t)lds_dst;  // low 32 bits = LDS offset
  asm volatile("global_load_async_to_lds_b128 %0, %1, off"
               :: "v"(lds_addr), "v"(gsrc)
               : "memory");
}
__device__ __forceinline__ void wait_async0() {
#if __has_builtin(__builtin_amdgcn_s_wait_asynccnt)
  __builtin_amdgcn_s_wait_asynccnt(0);
#else
  asm volatile("s_wait_asynccnt 0x0" ::: "memory");
#endif
}

// ---- A fragment (bf16 16xK=32), ISA 7.12.2 layout:
// lanes 0-15: V0..V3 hold K=0..7, V4..V7 hold K=16..23
// lanes16-31: V0..V3 hold K=8..15, V4..V7 hold K=24..31
__device__ __forceinline__ v16bf load_a_frag(const unsigned short* p, int rowStride,
                                             int colStride, int lane) {
  int m = lane & 15, kh = lane >> 4;
  us16 u;
#pragma unroll
  for (int j = 0; j < 8; ++j) {
    int kb = (j < 4) ? (2 * j) : (16 + 2 * (j - 4));
    int k  = kb + 8 * kh;
    if (colStride == 1) {
      unsigned int w = *(const unsigned int*)(p + m * rowStride + k);
      u[2 * j]     = (unsigned short)(w & 0xffffu);
      u[2 * j + 1] = (unsigned short)(w >> 16);
    } else {
      u[2 * j]     = p[m * rowStride + k * colStride];
      u[2 * j + 1] = p[m * rowStride + (k + 1) * colStride];
    }
  }
  return __builtin_bit_cast(v16bf, u);
}

// ---- B fragment (bf16 K=32 x 16), source stored TRANSPOSED as [n][k]:
// lanes 0-15: V0..V3 hold K=0..7, V4..V7 hold K=16..23; lanes 16-31: K=8..15, K=24..31
__device__ __forceinline__ v16bf load_bT_frag(const unsigned short* p, int rowStride, int lane) {
  int n = lane & 15, kh = lane >> 4;
  us16 u;
#pragma unroll
  for (int j = 0; j < 8; ++j) {
    int k = ((j < 4) ? (2 * j) : (16 + 2 * (j - 4))) + 8 * kh;
    unsigned int w = *(const unsigned int*)(p + n * rowStride + k);
    u[2 * j]     = (unsigned short)(w & 0xffffu);
    u[2 * j + 1] = (unsigned short)(w >> 16);
  }
  return __builtin_bit_cast(v16bf, u);
}

// ---- C/D fragment (f32 16x16): VGPR r -> M = r + 8*(lane>=16), N = lane&15
__device__ __forceinline__ v8f load_c_frag(const float* p, int stride, int lane) {
  int n = lane & 15, mh = lane >> 4;
  v8f c;
#pragma unroll
  for (int r = 0; r < 8; ++r) c[r] = p[(r + 8 * mh) * stride + n];
  return c;
}
__device__ __forceinline__ void store_c_frag(float* p, int stride, int lane, v8f c) {
  int n = lane & 15, mh = lane >> 4;
#pragma unroll
  for (int r = 0; r < 8; ++r) p[(r + 8 * mh) * stride + n] = c[r];
}

// ============================================================================
// Kernel 1: QKV projection. M=16384, N=1536, K=512. Tile 128x128, 256 thr.
// Writes Q/K/V bf16 in [b][h][l][d] layout (bias fused).
// ============================================================================
__global__ __launch_bounds__(256)
void qkv_gemm(const float* __restrict__ x, const float* __restrict__ w,
              const float* __restrict__ bias,
              unsigned short* __restrict__ Qb, unsigned short* __restrict__ Kb,
              unsigned short* __restrict__ Vb) {
  __shared__ __align__(16) unsigned short As[128 * 40];  // [m][k], stride 40
  __shared__ __align__(16) unsigned short Bt[128 * 40];  // [n][k], stride 40
  const int tid  = threadIdx.x;
  const int lane = tid & 31, wave = tid >> 5;
  const int m0 = blockIdx.x * 128;
  const int n0 = blockIdx.y * 128;

  v8f acc[8] = {};
  for (int k0 = 0; k0 < 512; k0 += 32) {
    for (int i = tid; i < 128 * 32; i += 256) {
      int r = i >> 5, c = i & 31;
      As[r * 40 + c] = f2bf(x[(size_t)(m0 + r) * 512 + (k0 + c)]);
    }
    for (int i = tid; i < 32 * 128; i += 256) {
      int kk = i >> 7, nn = i & 127;
      Bt[nn * 40 + kk] = f2bf(w[(size_t)(k0 + kk) * 1536 + (n0 + nn)]);
    }
    __syncthreads();
    v16bf a = load_a_frag(As + 16 * wave * 40, 40, 1, lane);
#pragma unroll
    for (int nf = 0; nf < 8; ++nf) {
      v16bf bm = load_bT_frag(Bt + nf * 16 * 40, 40, lane);
      acc[nf] = WMMA_BF16(a, bm, acc[nf]);
    }
    __syncthreads();
  }
  // epilogue: bias + scatter. which (q/k/v) is uniform per block: 512 % 128 == 0
  const int which = n0 >> 9;
  unsigned short* __restrict__ dst = (which == 0) ? Qb : ((which == 1) ? Kb : Vb);
  const int n = lane & 15, mh = lane >> 4;
#pragma unroll
  for (int nf = 0; nf < 8; ++nf) {
#pragma unroll
    for (int r = 0; r < 8; ++r) {
      int gm = m0 + 16 * wave + r + 8 * mh;
      int gn = n0 + nf * 16 + n;
      float v = acc[nf][r] + bias[gn];
      int bidx = gm >> 11, l = gm & 2047;
      int h = (gn >> 6) & 7, d = gn & 63;
      size_t off = (((size_t)bidx * NUM_HEADS + h) * SEQ + l) * HEAD_DIM + d;
      dst[off] = f2bf(v);
    }
  }
}

// ============================================================================
// Kernel 2: flash attention. One block = (b, h, 64-query tile). 256 threads.
// Q/K tiles staged with GLOBAL_LOAD_ASYNC_TO_LDS_B128 (ASYNCcnt path).
// ============================================================================
__global__ __launch_bounds__(256)
void attn_kernel(const unsigned short* __restrict__ Qb,
                 const unsigned short* __restrict__ Kb,
                 const unsigned short* __restrict__ Vb,
                 const unsigned char* __restrict__ mask,
                 const float* __restrict__ rel_table,
                 unsigned short* __restrict__ Ctx) {
  __shared__ __align__(16) unsigned short Qs[64 * 72];  // [q][d]
  __shared__ __align__(16) unsigned short Ks[64 * 72];  // [key][d]  (== B^T for QK^T)
  __shared__ __align__(16) unsigned short Vt[64 * 72];  // [d][key]  (== B^T for PV)
  __shared__ __align__(16) float Ss[64 * 66];           // scores f32; P bf16 aliased in low halves
  __shared__ __align__(16) float Os[64 * 66];           // output accumulator
  __shared__ float m_i[64], l_i[64], alf[64];
  __shared__ float red[64 * 4];
  __shared__ float relB[2 * MAX_DIST + 1];

  const int tid = threadIdx.x, lane = tid & 31, wave = tid >> 5;
  const int q0 = blockIdx.x * 64;
  const int h  = blockIdx.y;
  const int b  = blockIdx.z;
  const size_t headBase = ((size_t)b * NUM_HEADS + h) * SEQ * HEAD_DIM;
  const float scale = 0.125f;  // HEAD_DIM^-0.5

  for (int i = tid; i < 2 * MAX_DIST + 1; i += 256) relB[i] = rel_table[i * NUM_HEADS + h];
  for (int i = tid; i < 64; i += 256) { m_i[i] = -INFINITY; l_i[i] = 0.f; }
  for (int i = tid; i < 64 * 64; i += 256) Os[(i >> 6) * 66 + (i & 63)] = 0.f;

  // ---- Q tile: 64 rows x 128B = 512 x 16B chunks, async to LDS
  {
    const unsigned short* gq = Qb + headBase + (size_t)q0 * HEAD_DIM;
#pragma unroll
    for (int it = 0; it < 2; ++it) {
      int ch = tid + 256 * it;
      int r = ch >> 3, c8 = (ch & 7) * 8;      // 8 ushorts = 16B
      async_g2l_b128(Qs + r * 72 + c8, gq + (size_t)r * HEAD_DIM + c8);
    }
  }
  wait_async0();
  __syncthreads();

  for (int kt = 0; kt < SEQ / 64; ++kt) {
    const int k0 = kt * 64;
    // ---- K tile via async DMA to LDS
    {
      const unsigned short* gk = Kb + headBase + (size_t)k0 * HEAD_DIM;
#pragma unroll
      for (int it = 0; it < 2; ++it) {
        int ch = tid + 256 * it;
        int r = ch >> 3, c8 = (ch & 7) * 8;
        async_g2l_b128(Ks + r * 72 + c8, gk + (size_t)r * HEAD_DIM + c8);
      }
    }
    // ---- V tile transposed into LDS (manual: async cannot transpose)
    for (int i = tid; i < 64 * 64; i += 256) {
      int r = i >> 6, c = i & 63;                      // r = key, c = d
      Vt[c * 72 + r] = Vb[headBase + (size_t)(k0 + r) * HEAD_DIM + c];
    }
    wait_async0();
    __syncthreads();

    // ---- S = scale * Q K^T + rel_bias (+ mask). 16 frags, 2 per wave.
#pragma unroll
    for (int t2 = 0; t2 < 2; ++t2) {
      int t = wave * 2 + t2, tm = t >> 2, tn = t & 3;
      v8f c = {};
#pragma unroll
      for (int ks = 0; ks < 2; ++ks) {
        v16bf a  = load_a_frag(Qs + tm * 16 * 72 + ks * 32, 72, 1, lane);
        v16bf bm = load_bT_frag(Ks + tn * 16 * 72 + ks * 32, 72, lane);
        c = WMMA_BF16(a, bm, c);
      }
      int n = lane & 15, mh = lane >> 4;
      int ncol = tn * 16 + n;
      int kpos = k0 + ncol;
      bool ok  = mask[(size_t)b * SEQ + kpos] != 0;
#pragma unroll
      for (int r = 0; r < 8; ++r) {
        int m = tm * 16 + r + 8 * mh;
        int rel = kpos - (q0 + m);
        rel = rel < -MAX_DIST ? -MAX_DIST : (rel > MAX_DIST ? MAX_DIST : rel);
        float s = c[r] * scale + relB[rel + MAX_DIST];
        Ss[m * 66 + ncol] = ok ? s : -INFINITY;
      }
    }
    __syncthreads();

    // ---- online softmax: row max
    {
      int r = tid >> 2, seg = tid & 3;
      float mx = -INFINITY;
#pragma unroll
      for (int c = 0; c < 16; ++c) mx = fmaxf(mx, Ss[r * 66 + seg * 16 + c]);
      red[r * 4 + seg] = mx;
    }
    __syncthreads();
    if (tid < 64) {
      float mx = fmaxf(fmaxf(red[tid * 4 + 0], red[tid * 4 + 1]),
                       fmaxf(red[tid * 4 + 2], red[tid * 4 + 3]));
      float mo = m_i[tid];
      float mn = fmaxf(mo, mx);
      alf[tid] = __expf(mo - mn);   // exp(-inf)=0 on first tile
      m_i[tid] = mn;
    }
    __syncthreads();
    // ---- P = exp(S-m) (bf16 in place), rescale O, partial row sums
    {
      int r = tid >> 2, seg = tid & 3;
      float mn = m_i[r], al = alf[r], sum = 0.f;
#pragma unroll
      for (int c = 0; c < 16; ++c) {
        int cc = seg * 16 + c;
        float p = __expf(Ss[r * 66 + cc] - mn);
        sum += p;
        ((unsigned short*)&Ss[r * 66 + cc])[0] = f2bf(p);
        Os[r * 66 + cc] *= al;
      }
      red[r * 4 + seg] = sum;
    }
    __syncthreads();
    if (tid < 64)
      l_i[tid] = l_i[tid] * alf[tid] +
                 red[tid * 4] + red[tid * 4 + 1] + red[tid * 4 + 2] + red[tid * 4 + 3];
    __syncthreads();

    // ---- O += P @ V. 16 frags, 2 per wave, C from/to LDS accumulator.
#pragma unroll
    for (int t2 = 0; t2 < 2; ++t2) {
      int t = wave * 2 + t2, tm = t >> 2, tn = t & 3;
      v8f c = load_c_frag(Os + tm * 16 * 66 + tn * 16, 66, lane);
#pragma unroll
      for (int ks = 0; ks < 2; ++ks) {
        // P: bf16 in low half of each f32 S cell -> rowStride 132 ushorts, colStride 2
        v16bf a  = load_a_frag((const unsigned short*)Ss + tm * 16 * 132 + ks * 64, 132, 2, lane);
        v16bf bm = load_bT_frag(Vt + tn * 16 * 72 + ks * 32, 72, lane);
        c = WMMA_BF16(a, bm, c);
      }
      store_c_frag(Os + tm * 16 * 66 + tn * 16, 66, lane, c);
    }
    __syncthreads();
  }

  // ---- normalize and write context bf16: Ctx[b][l][h*64+d]
  {
    int r = tid >> 2, seg = tid & 3;
    float inv = 1.0f / l_i[r];
    size_t base = ((size_t)b * SEQ + (q0 + r)) * D_MODEL + h * HEAD_DIM + seg * 16;
#pragma unroll
    for (int c = 0; c < 16; ++c)
      Ctx[base + c] = f2bf(Os[r * 66 + seg * 16 + c] * inv);
  }
}

// ============================================================================
// Kernel 3: output projection. M=16384, N=512, K=512. Tile 128x128, fused b_out.
// ============================================================================
__global__ __launch_bounds__(256)
void out_gemm(const unsigned short* __restrict__ Ctx, const float* __restrict__ w,
              const float* __restrict__ bias, float* __restrict__ out) {
  __shared__ __align__(16) unsigned short As[128 * 40];
  __shared__ __align__(16) unsigned short Bt[128 * 40];
  const int tid = threadIdx.x, lane = tid & 31, wave = tid >> 5;
  const int m0 = blockIdx.x * 128, n0 = blockIdx.y * 128;

  v8f acc[8] = {};
  for (int k0 = 0; k0 < 512; k0 += 32) {
    for (int i = tid; i < 128 * 32; i += 256) {
      int r = i >> 5, c = i & 31;
      As[r * 40 + c] = Ctx[(size_t)(m0 + r) * 512 + (k0 + c)];
    }
    for (int i = tid; i < 32 * 128; i += 256) {
      int kk = i >> 7, nn = i & 127;
      Bt[nn * 40 + kk] = f2bf(w[(size_t)(k0 + kk) * 512 + (n0 + nn)]);
    }
    __syncthreads();
    v16bf a = load_a_frag(As + 16 * wave * 40, 40, 1, lane);
#pragma unroll
    for (int nf = 0; nf < 8; ++nf) {
      v16bf bm = load_bT_frag(Bt + nf * 16 * 40, 40, lane);
      acc[nf] = WMMA_BF16(a, bm, acc[nf]);
    }
    __syncthreads();
  }
  const int n = lane & 15, mh = lane >> 4;
#pragma unroll
  for (int nf = 0; nf < 8; ++nf) {
#pragma unroll
    for (int r = 0; r < 8; ++r) {
      int gm = m0 + 16 * wave + r + 8 * mh;
      int gn = n0 + nf * 16 + n;
      out[(size_t)gm * 512 + gn] = acc[nf][r] + bias[gn];
    }
  }
}

// ============================================================================
extern "C" void kernel_launch(void* const* d_in, const int* in_sizes, int n_in,
                              void* d_out, int out_size, void* d_ws, size_t ws_size,
                              hipStream_t stream) {
  const float*         x         = (const float*)d_in[0];
  const unsigned char* mask      = (const unsigned char*)d_in[1];
  const float*         w_qkv     = (const float*)d_in[2];
  const float*         b_qkv     = (const float*)d_in[3];
  const float*         w_out     = (const float*)d_in[4];
  const float*         b_out     = (const float*)d_in[5];
  const float*         rel_table = (const float*)d_in[6];
  float*               out       = (float*)d_out;

  const size_t headElems = (size_t)BATCH * NUM_HEADS * SEQ * HEAD_DIM;  // 8,388,608
  unsigned short* Qb  = (unsigned short*)d_ws;
  unsigned short* Kb  = Qb + headElems;
  unsigned short* Vb  = Kb + headElems;
  unsigned short* Ctx = Vb + headElems;   // [B][L][D_MODEL] bf16

  dim3 g1(NROW / 128, (3 * D_MODEL) / 128);
  qkv_gemm<<<g1, 256, 0, stream>>>(x, w_qkv, b_qkv, Qb, Kb, Vb);

  dim3 g2(SEQ / 64, NUM_HEADS, BATCH);
  attn_kernel<<<g2, 256, 0, stream>>>(Qb, Kb, Vb, mask, rel_table, Ctx);

  dim3 g3(NROW / 128, D_MODEL / 128);
  out_gemm<<<g3, 256, 0, stream>>>(Ctx, w_out, b_out, out);
}
